// MultiHeadAttention_76149770158442
// MI455X (gfx1250) — matrix operand order repaired and checked by reading, not compiled, not run on hardware
//
#include <hip/hip_runtime.h>
#include <hip/hip_bf16.h>
#include <stdint.h>

// ---------------------------------------------------------------------------
// MHA forward for MI455X (gfx1250, wave32, WMMA).
// Heavy math: v_wmma_f32_16x16x32_bf16 (bf16 in, f32 accumulate).
// K/V attention tiles: GLOBAL_LOAD_ASYNC_TO_LDS_B128 (ASYNCcnt path).
// ---------------------------------------------------------------------------

#define DEVI __device__ __forceinline__

typedef __attribute__((ext_vector_type(16))) __bf16 v16bf;
typedef __attribute__((ext_vector_type(8)))  __bf16 v8bf;
typedef __attribute__((ext_vector_type(8)))  float  v8f;

// Problem constants (fixed by the reference).
constexpr int Bsz = 2;
constexpr int Sl  = 2048;
constexpr int Din = 1024;
constexpr int Hh  = 16;
constexpr int DKk = 64;
constexpr int Mg  = Bsz * Sl;   // 4096 rows for every GEMM
constexpr int Kg  = 1024;       // K dim of every GEMM
constexpr int Ng  = 1024;       // N dim of every GEMM (H*DK == Din)

DEVI __bf16 f2bf(float f) {
  union { float f; uint32_t u; } v; v.f = f;
  uint32_t r = (v.u + 0x7FFFu + ((v.u >> 16) & 1u)) >> 16;  // RNE
  uint16_t h = (uint16_t)r;
  __bf16 b;
  __builtin_memcpy(&b, &h, 2);
  return b;
}

DEVI v16bf cat8(v8bf lo, v8bf hi) {
  return __builtin_shufflevector(lo, hi, 0, 1, 2, 3, 4, 5, 6, 7,
                                         8, 9, 10, 11, 12, 13, 14, 15);
}

DEVI v8f wmma_bf16(v16bf a, v16bf b, v8f c) {
  // (neg_a, A, neg_b, B, c_mod, C, reuse_a, reuse_b)
  return __builtin_amdgcn_wmma_f32_16x16x32_bf16(false, a, false, b,
                                                 (short)0, c, false, false);
}

// Generic LDS pointer -> 32-bit wave-relative LDS byte offset (addr[31:0]).
DEVI uint32_t lds_off32(const void* p) { return (uint32_t)(uintptr_t)p; }

// Async global -> LDS copy, 16 bytes per lane, tracked by ASYNCcnt.
DEVI void async_g2l_b128(uint32_t lds_byte_off, const void* gptr) {
  asm volatile("global_load_async_to_lds_b128 %0, %1, off"
               :
               : "v"(lds_byte_off), "v"((uint64_t)(uintptr_t)gptr)
               : "memory");
}

DEVI void wait_asynccnt0() {
  asm volatile("s_wait_asynccnt 0x0" ::: "memory");
}

// ---------------------------------------------------------------------------
// fp32 -> bf16 elementwise cast
// ---------------------------------------------------------------------------
__global__ __launch_bounds__(256) void k_cast_bf16(const float* __restrict__ in,
                                                   __bf16* __restrict__ out,
                                                   int n) {
  int i = blockIdx.x * 256 + threadIdx.x;
  if (i < n) out[i] = f2bf(in[i]);
}

// Repack per-head weight W[h][d][k]  ->  Wc[d][h*64 + k]  (bf16).
__global__ __launch_bounds__(256) void k_pack_w(const float* __restrict__ in,
                                                __bf16* __restrict__ out,
                                                int n) {
  int i = blockIdx.x * 256 + threadIdx.x;
  if (i < n) {
    int k = i & 63;
    int d = (i >> 6) & (Din - 1);
    int h = i >> 16;                       // 64*1024 per head
    out[(size_t)d * Ng + h * DKk + k] = f2bf(in[i]);
  }
}

// ---------------------------------------------------------------------------
// WMMA GEMM: C = A(4096x1024) * B(1024x1024) + bias[col]
//   MODE 0: store bf16 as Q/V head layout [B,H,S,DK]
//   MODE 1: store bf16 as Kt head layout  [B,H,DK,S]  (pre-transposed for attn)
//   MODE 2: store fp32 row-major [M,N]    (final output projection)
// Block: 256 thr (8 waves), 128x128 tile; wave tile 32x64 (2x4 WMMA accums).
// ---------------------------------------------------------------------------
template <int MODE>
__global__ __launch_bounds__(256)
void k_gemm4096(const __bf16* __restrict__ A, const __bf16* __restrict__ Bm,
                const float* __restrict__ bias, void* __restrict__ outp) {
  const int tid = threadIdx.x;
  const int w = tid >> 5, l = tid & 31;
  const int lane16 = l & 15, hi = l >> 4, kbase = hi * 8;
  const int m_base = blockIdx.y * 128 + (w & 3) * 32;
  const int n_base = blockIdx.x * 128 + (w >> 2) * 64;

  const v8f z = {};
  v8f acc[2][4];
#pragma unroll
  for (int i = 0; i < 2; ++i)
#pragma unroll
    for (int j = 0; j < 4; ++j) acc[i][j] = z;

  for (int k0 = 0; k0 < Kg; k0 += 32) {
    // A fragment: lane holds row m_base+ms*16+lane16; K elems at
    // [k0+kbase .. +7] and [k0+kbase+16 .. +23] per the 16-bit A VGPR layout.
    v16bf af[2];
#pragma unroll
    for (int ms = 0; ms < 2; ++ms) {
      const __bf16* ap =
          A + (size_t)(m_base + ms * 16 + lane16) * Kg + k0 + kbase;
      af[ms] = cat8(*(const v8bf*)ap, *(const v8bf*)(ap + 16));
    }
    // B fragment: lane = N col (lane16), K elems kbase+s (s<8), kbase+8+s (s>=8).
    const __bf16* bp = Bm + (size_t)(k0 + kbase) * Ng + n_base + lane16;
    __builtin_prefetch(bp + 32 * Ng, 0, 1);  // next K tile -> global_prefetch
    v16bf bfr[4];
#pragma unroll
    for (int ns = 0; ns < 4; ++ns)
#pragma unroll
      for (int s = 0; s < 16; ++s)
        bfr[ns][s] = bp[(size_t)(s + (s & 8)) * Ng + ns * 16];
#pragma unroll
    for (int ms = 0; ms < 2; ++ms)
#pragma unroll
      for (int ns = 0; ns < 4; ++ns)
        acc[ms][ns] = wmma_bf16(af[ms], bfr[ns], acc[ms][ns]);
  }

  // Epilogue. C layout: lane -> N = n_base+ns*16+lane16, VGPR r -> M offset
  // r + 8*hi within the 16-row tile.
#pragma unroll
  for (int ms = 0; ms < 2; ++ms) {
#pragma unroll
    for (int ns = 0; ns < 4; ++ns) {
      const int n = n_base + ns * 16 + lane16;
      const float bv = bias[n];
#pragma unroll
      for (int r = 0; r < 8; ++r) {
        const int m = m_base + ms * 16 + r + hi * 8;
        const float val = acc[ms][ns][r] + bv;
        if constexpr (MODE == 2) {
          ((float*)outp)[(size_t)m * Ng + n] = val;
        } else {
          const int b = m >> 11, s = m & (Sl - 1);
          const int h = n >> 6, kk = n & 63;
          const size_t idx =
              (MODE == 0)
                  ? ((((size_t)b * Hh + h) * Sl + s) * DKk + kk)   // [B,H,S,DK]
                  : ((((size_t)b * Hh + h) * DKk + kk) * Sl + s);  // [B,H,DK,S]
          ((__bf16*)outp)[idx] = f2bf(val);
        }
      }
    }
  }
}

// ---------------------------------------------------------------------------
// Flash attention per (b, h, 128-query block). 8 waves, 16 query rows each.
// K/V key-tiles (32 keys) staged in LDS via async global->LDS b128 copies
// (ASYNCcnt), shared by all waves; P tile round-trips through per-wave LDS
// to convert C-layout -> A-layout.
// ---------------------------------------------------------------------------
__global__ __launch_bounds__(256)
void k_flash_attn(const __bf16* __restrict__ Q,   // [B,H,S,DK]
                  const __bf16* __restrict__ Kt,  // [B,H,DK,S]
                  const __bf16* __restrict__ V,   // [B,H,S,DK]
                  __bf16* __restrict__ O) {       // [B*S, H*DK]
  __shared__ __attribute__((aligned(16))) __bf16 sK[64 * 32];      // [d][t]
  __shared__ __attribute__((aligned(16))) __bf16 sV[32 * 64];      // [t][d]
  __shared__ __attribute__((aligned(16))) __bf16 sP[8][16 * 32];   // per wave

  const int tid = threadIdx.x;
  const int w = tid >> 5, l = tid & 31;
  const int lane16 = l & 15, hi = l >> 4, kbase = hi * 8;
  const int h = blockIdx.y, b = blockIdx.z;
  const int q0 = blockIdx.x * 128 + w * 16;

  const size_t head = (size_t)b * Hh + h;
  const __bf16* Qh = Q + head * Sl * DKk;   // [S][64]
  const __bf16* Kh = Kt + head * DKk * Sl;  // [64][S]
  const __bf16* Vh = V + head * Sl * DKk;   // [S][64]

  // Per-thread async-copy coordinates (16B per thread per tile).
  const int kd = tid >> 2, kc = (tid & 3) * 8;   // sK: 64 x 32
  const int vt = tid >> 3, vc = (tid & 7) * 8;   // sV: 32 x 64
  const uint32_t sK_dst = lds_off32(sK + kd * 32 + kc);
  const uint32_t sV_dst = lds_off32(sV + vt * 64 + vc);

  // Q fragments for this wave's 16 rows (two K-chunks over DK=64), kept live.
  v16bf qf[2];
#pragma unroll
  for (int c = 0; c < 2; ++c) {
    const __bf16* qp = Qh + (size_t)(q0 + lane16) * DKk + c * 32 + kbase;
    qf[c] = cat8(*(const v8bf*)qp, *(const v8bf*)(qp + 16));
  }

  const v8f z = {};
  v8f accO[4];
#pragma unroll
  for (int i = 0; i < 4; ++i) accO[i] = z;
  float mrow[8], lrow[8];
#pragma unroll
  for (int r = 0; r < 8; ++r) { mrow[r] = -1e30f; lrow[r] = 0.0f; }

  for (int t0 = 0; t0 < Sl; t0 += 32) {
    __syncthreads();  // previous iteration done with sK/sV
    // Async tile fill: VGPR-free global->LDS, tracked by ASYNCcnt.
    async_g2l_b128(sK_dst, Kh + (size_t)kd * Sl + t0 + kc);
    async_g2l_b128(sV_dst, Vh + (size_t)(t0 + vt) * DKk + vc);
    wait_asynccnt0();   // this wave's copies landed in LDS
    __syncthreads();    // everyone's copies visible

    // scores = Q * K^T for 32 keys -> two 16x16 f32 accumulators
    v8f sc[2] = {z, z};
#pragma unroll
    for (int c = 0; c < 2; ++c) {
      const __bf16* kp = sK + (c * 32 + kbase) * 32 + lane16;  // lane = key col
#pragma unroll
      for (int ts = 0; ts < 2; ++ts) {
        v16bf bk;
#pragma unroll
        for (int s = 0; s < 16; ++s) bk[s] = kp[(s + (s & 8)) * 32 + ts * 16];
        sc[ts] = wmma_bf16(qf[c], bk, sc[ts]);
      }
    }

    // Online softmax. Row r lives in VGPR r across the 16 lanes of each half.
#pragma unroll
    for (int r = 0; r < 8; ++r) {
      const float s0 = sc[0][r] * 0.125f;  // 1/sqrt(DK)
      const float s1 = sc[1][r] * 0.125f;
      float mx = fmaxf(s0, s1);
#pragma unroll
      for (int d = 1; d < 16; d <<= 1) mx = fmaxf(mx, __shfl_xor(mx, d, 32));
      const float mnew = fmaxf(mrow[r], mx);
      const float alpha = __expf(mrow[r] - mnew);
      const float p0 = __expf(s0 - mnew);
      const float p1 = __expf(s1 - mnew);
      float ps = p0 + p1;
#pragma unroll
      for (int d = 1; d < 16; d <<= 1) ps += __shfl_xor(ps, d, 32);
      lrow[r] = lrow[r] * alpha + ps;
      mrow[r] = mnew;
#pragma unroll
      for (int ns = 0; ns < 4; ++ns) accO[ns][r] *= alpha;
      const int prow = r + hi * 8;
      sP[w][prow * 32 + lane16]      = f2bf(p0);
      sP[w][prow * 32 + 16 + lane16] = f2bf(p1);
    }

    // Wave-local LDS RAW (C-layout stores -> A-layout loads), same wave only.
    asm volatile("s_wait_dscnt 0x0" ::: "memory");

    // P as A fragment (16x32), V tile as B fragments from LDS.
    const __bf16* pp = &sP[w][lane16 * 32 + kbase];
    v16bf pf = cat8(*(const v8bf*)pp, *(const v8bf*)(pp + 16));
#pragma unroll
    for (int ns = 0; ns < 4; ++ns) {
      const __bf16* vp = sV + kbase * 64 + ns * 16 + lane16;  // lane = d col
      v16bf bv;
#pragma unroll
      for (int s = 0; s < 16; ++s) bv[s] = vp[(s + (s & 8)) * 64];
      accO[ns] = wmma_bf16(pf, bv, accO[ns]);
    }
  }

  // Normalize and store O[b*S+s][h*64+col] in bf16.
  float linv[8];
#pragma unroll
  for (int r = 0; r < 8; ++r) linv[r] = 1.0f / lrow[r];
#pragma unroll
  for (int ns = 0; ns < 4; ++ns) {
    const int col = h * DKk + ns * 16 + lane16;
#pragma unroll
    for (int r = 0; r < 8; ++r) {
      const float o = accO[ns][r] * linv[r];
      const int s = q0 + r + hi * 8;
      O[(size_t)(b * Sl + s) * Ng + col] = f2bf(o);
    }
  }
}

// ---------------------------------------------------------------------------
// Launcher
// ---------------------------------------------------------------------------
extern "C" void kernel_launch(void* const* d_in, const int* in_sizes, int n_in,
                              void* d_out, int out_size, void* d_ws,
                              size_t ws_size, hipStream_t stream) {
  (void)in_sizes; (void)n_in; (void)out_size; (void)ws_size;

  const float* q  = (const float*)d_in[0];
  const float* k  = (const float*)d_in[1];
  const float* v  = (const float*)d_in[2];
  const float* Wq = (const float*)d_in[3];
  const float* bq = (const float*)d_in[4];
  const float* Wk = (const float*)d_in[5];
  const float* bk = (const float*)d_in[6];
  const float* Wv = (const float*)d_in[7];
  const float* bv = (const float*)d_in[8];
  const float* Wo = (const float*)d_in[9];
  const float* bo = (const float*)d_in[10];

  // Workspace layout (bf16, 256B aligned segments): ~64 MB total.
  char* ws = (char*)d_ws;
  size_t off = 0;
  auto seg = [&](size_t elems) {
    __bf16* p = (__bf16*)(ws + off);
    off += (elems * sizeof(__bf16) + 255) & ~(size_t)255;
    return p;
  };
  const size_t NX = (size_t)Mg * Kg;      // 4096*1024
  const size_t NW = (size_t)Kg * Ng;      // 1024*1024
  __bf16* Xq  = seg(NX);
  __bf16* Xk  = seg(NX);
  __bf16* Xv  = seg(NX);
  __bf16* Wqc = seg(NW);
  __bf16* Wkc = seg(NW);
  __bf16* Wvc = seg(NW);
  __bf16* Woc = seg(NW);
  __bf16* Qb  = seg(NX);
  __bf16* Ktb = seg(NX);
  __bf16* Vb  = seg(NX);
  __bf16* Ob  = seg(NX);

  const int nX = (int)NX, nW = (int)NW;
  const dim3 blk(256);
  // Casts / packs
  k_cast_bf16<<<(nX + 255) / 256, blk, 0, stream>>>(q, Xq, nX);
  k_cast_bf16<<<(nX + 255) / 256, blk, 0, stream>>>(k, Xk, nX);
  k_cast_bf16<<<(nX + 255) / 256, blk, 0, stream>>>(v, Xv, nX);
  k_cast_bf16<<<(nW + 255) / 256, blk, 0, stream>>>(Wo, Woc, nW);
  k_pack_w<<<(nW + 255) / 256, blk, 0, stream>>>(Wq, Wqc, nW);
  k_pack_w<<<(nW + 255) / 256, blk, 0, stream>>>(Wk, Wkc, nW);
  k_pack_w<<<(nW + 255) / 256, blk, 0, stream>>>(Wv, Wvc, nW);

  // QKV projections (all 4096x1024x1024 WMMA GEMMs)
  const dim3 ggrid(Ng / 128, Mg / 128);  // (8, 32)
  k_gemm4096<0><<<ggrid, blk, 0, stream>>>(Xq, Wqc, bq, Qb);
  k_gemm4096<1><<<ggrid, blk, 0, stream>>>(Xk, Wkc, bk, Ktb);
  k_gemm4096<0><<<ggrid, blk, 0, stream>>>(Xv, Wvc, bv, Vb);

  // Flash attention
  const dim3 agrid(Sl / 128, Hh, Bsz);   // (16, 16, 2)
  k_flash_attn<<<agrid, blk, 0, stream>>>(Qb, Ktb, Vb, Ob);

  // Output projection -> fp32 d_out
  k_gemm4096<2><<<ggrid, blk, 0, stream>>>(Ob, Woc, bo, d_out);
}
